// ReasoningLayer_8555574854148
// MI455X (gfx1250) — matrix operand order repaired and checked by reading, not compiled, not run on hardware
//
#include <hip/hip_runtime.h>
#include <hip/hip_bf16.h>
#include <math.h>

// ---------------------------------------------------------------------------
// Fused per-token MLP: out = GELU(X*W1 + b1)*W2 + b2
//   X: [16384, 1024] f32, W1: [1024, 2048] f32, W2: [2048, 1024] f32
// MI455X (gfx1250, wave32) strategy:
//   - pre-pass swizzles W1/W2 -> bf16 WMMA-B-fragment layout in d_ws (8 MB):
//     hot-loop B loads become 4 coalesced global_load_b128 per k-step
//     (vs 32 strided b32 gathers), and L2 weight traffic halves
//   - one workgroup (256 thr = 8 waves) per 64-token M-tile (grid = 256)
//   - GEMM1 -> branch-free tanh-GELU -> bf16 H-tile in LDS (256 KB of 320 KB)
//   - GEMM2 out of LDS, f32 result to HBM
//   - per-wave register tile: 2 n-tiles x 4 m-subtiles -> 8 WMMAs per
//     B-fragment pair (B reuse x4, A reuse x2)
//   - all matrix math via v_wmma_f32_16x16x32_bf16 (f32 accumulate)
// ---------------------------------------------------------------------------

typedef __bf16 bf16_t;
typedef __attribute__((ext_vector_type(16))) __bf16 v16bf;
typedef __attribute__((ext_vector_type(8)))  __bf16 v8bf;
typedef __attribute__((ext_vector_type(8)))  float  v8f;

#define D_DIM 1024
#define H_DIM 2048
#define S_DIM 2048
#define B_DIM 8
#define M_TOT (B_DIM * S_DIM)            // 16384 tokens
#define MTILE 64                          // rows per workgroup
#define LDS_BYTES (MTILE * H_DIM * 2)    // 262144
#define W1_ELEMS (D_DIM * H_DIM)         // 2M
#define W2_ELEMS (H_DIM * D_DIM)         // 2M
#define WS_NEEDED ((size_t)(W1_ELEMS + W2_ELEMS) * 2)   // 8 MB bf16 weights
#define SWZ1_BLOCKS (32 * 128)  // W1: kblk(0..31) major, nt(0..127) minor
#define SWZ2_BLOCKS (64 * 64)   // W2: j(0..63) major,    nt(0..63)  minor

__device__ __forceinline__ v8f wmma_bf16(v16bf a, v16bf b, v8f c) {
    return __builtin_amdgcn_wmma_f32_16x16x32_bf16(false, a, false, b,
                                                   (short)0, c, false, false);
}

// tanh-form GELU, branch-free: x * sigmoid(1.5957691*x*(1+0.044715x^2)).
// v_exp_f32 saturates to +inf/0 -> v_rcp_f32 gives exact 0/1 tails, no branches.
__device__ __forceinline__ float gelu_fast(float x) {
    float u = x * (0.7978845608028654f + 0.03567740814f * x * x);
    float e = __builtin_amdgcn_exp2f(-2.8853900817779268f * u); // exp(-2u)
    return x * __builtin_amdgcn_rcpf(1.0f + e);
}

// ---------------------------------------------------------------------------
// Weight swizzle pre-pass: f32 row-major -> bf16 WMMA B-fragment blocks.
// Block (kblk, nt) = 512 bf16 (1 KB); lane L owns elems [L*16, L*16+16):
//   elem i = W[kblk*32 + 16*(L>>4) + i][nt*16 + (L&15)]
// One thread produces one lane-chunk (32 B contiguous write).
// ---------------------------------------------------------------------------
__global__ __launch_bounds__(256)
void swizzle_weights_kernel(const float* __restrict__ W1,
                            const float* __restrict__ W2,
                            bf16_t* __restrict__ swz1,
                            bf16_t* __restrict__ swz2) {
    const int t    = blockIdx.x * blockDim.x + threadIdx.x;  // 0..262143
    const int blk  = t >> 5;
    const int lane = t & 31;
    const int li   = lane & 15;
    const int lh   = lane >> 4;

    bf16_t* dst;
    const float* src;
    int stride;
    if (blk < SWZ1_BLOCKS) {
        const int kblk = blk >> 7;          // 0..31
        const int nt   = blk & 127;         // 0..127
        dst    = swz1 + (size_t)blk * 512 + lane * 16;
        src    = W1 + (size_t)(kblk * 32 + 16 * lh) * H_DIM + nt * 16 + li;
        stride = H_DIM;
    } else {
        const int b2 = blk - SWZ1_BLOCKS;
        const int j  = b2 >> 6;             // 0..63
        const int nt = b2 & 63;             // 0..63
        dst    = swz2 + (size_t)b2 * 512 + lane * 16;
        src    = W2 + (size_t)(j * 32 + 16 * lh) * D_DIM + nt * 16 + li;
        stride = D_DIM;
    }
    v8bf o0, o1;
    #pragma unroll
    for (int i = 0; i < 8; ++i) o0[i] = (bf16_t)src[(size_t)i * stride];
    #pragma unroll
    for (int i = 0; i < 8; ++i) o1[i] = (bf16_t)src[(size_t)(i + 8) * stride];
    *(v8bf*)(dst)     = o0;
    *(v8bf*)(dst + 8) = o1;
}

// ---------------------------------------------------------------------------
// Fused MLP kernel. SWZ=true: B from swizzled bf16 blocks (fast path);
// SWZ=false: B gathered from f32 weights (fallback).
// ---------------------------------------------------------------------------
template <bool SWZ>
__global__ __launch_bounds__(256)
void mlp_fused_kernel(const float* __restrict__ X,
                      const float* __restrict__ W1f,
                      const float* __restrict__ b1,
                      const float* __restrict__ W2f,
                      const float* __restrict__ b2,
                      const bf16_t* __restrict__ swz1,
                      const bf16_t* __restrict__ swz2,
                      float* __restrict__ Out) {
    // H tile: 128 n-tile blocks of [64 rows x 16 cols] bf16 (2 KB each) = 256 KB
    extern __shared__ bf16_t sH[];

    const int tid  = threadIdx.x;
    const int wave = tid >> 5;
    const int lane = tid & 31;
    const int col  = lane & 15;   // N-col (C/D layout) == A-row (A layout)
    const int hi   = lane >> 4;   // half-wave select
    const int m_base = blockIdx.x * MTILE;

    const float* Xrow[4];
    #pragma unroll
    for (int ms = 0; ms < 4; ++ms)
        Xrow[ms] = X + (size_t)(m_base + 16 * ms + col) * D_DIM;

    // ------------------- Phase 1: H = GELU(X*W1 + b1) -----------------------
    for (int ng = 0; ng < 8; ++ng) {
        const int ntb = 16 * ng + 2 * wave;
        const int n0  = ntb * 16 + col;
        const int n1  = n0 + 16;

        v8f acc[2][4];
        const float bia0 = b1[n0], bia1 = b1[n1];
        #pragma unroll
        for (int ms = 0; ms < 4; ++ms)
            #pragma unroll
            for (int r = 0; r < 8; ++r) { acc[0][ms][r] = bia0; acc[1][ms][r] = bia1; }

        for (int k0 = 0; k0 < D_DIM; k0 += 32) {
            // A fragments (16x32 bf16): lanes 0-15 K {k0..+15},
            // lanes 16-31 K {k0+16..+31}
            v16bf a[4];
            #pragma unroll
            for (int ms = 0; ms < 4; ++ms) {
                float af[16];
                *(float4*)(af + 0)  = *(const float4*)(Xrow[ms] + k0 + 16 * hi);
                *(float4*)(af + 4)  = *(const float4*)(Xrow[ms] + k0 + 16 * hi + 4);
                *(float4*)(af + 8)  = *(const float4*)(Xrow[ms] + k0 + 16 * hi + 8);
                *(float4*)(af + 12) = *(const float4*)(Xrow[ms] + k0 + 16 * hi + 12);
                #pragma unroll
                for (int i = 0; i < 16; ++i) a[ms][i] = (bf16_t)af[i];
            }
            // B fragments (32x16 bf16 pair for n-tiles ntb, ntb+1)
            v16bf bf0, bf1;
            if (SWZ) {
                const size_t wb = ((size_t)(k0 >> 5) * 128 + ntb) * 512 + lane * 16;
                v8bf l0 = *(const v8bf*)(swz1 + wb);
                v8bf h0 = *(const v8bf*)(swz1 + wb + 8);
                v8bf l1 = *(const v8bf*)(swz1 + wb + 512);
                v8bf h1 = *(const v8bf*)(swz1 + wb + 520);
                #pragma unroll
                for (int i = 0; i < 8; ++i) {
                    bf0[i] = l0[i]; bf0[i + 8] = h0[i];
                    bf1[i] = l1[i]; bf1[i + 8] = h1[i];
                }
            } else {
                const int kb = k0 + 16 * hi;
                #pragma unroll
                for (int i = 0; i < 16; ++i) {
                    const float* wr = W1f + (size_t)(kb + i) * H_DIM;
                    bf0[i] = (bf16_t)wr[n0];
                    bf1[i] = (bf16_t)wr[n1];
                }
            }
            #pragma unroll
            for (int ms = 0; ms < 4; ++ms) {
                acc[0][ms] = wmma_bf16(a[ms], bf0, acc[0][ms]);
                acc[1][ms] = wmma_bf16(a[ms], bf1, acc[1][ms]);
            }
        }

        // C/D layout: VGPR r -> M = 16*ms + r + 8*hi, N = col
        #pragma unroll
        for (int e = 0; e < 2; ++e)
            #pragma unroll
            for (int ms = 0; ms < 4; ++ms)
                #pragma unroll
                for (int r = 0; r < 8; ++r)
                    sH[(ntb + e) * (MTILE * 16) + (16 * ms + r + 8 * hi) * 16 + col] =
                        (bf16_t)gelu_fast(acc[e][ms][r]);
    }

    __syncthreads();

    // ------------------- Phase 2: out = H*W2 + b2 ---------------------------
    for (int ng = 0; ng < 4; ++ng) {
        const int ntb = 16 * ng + 2 * wave;
        const int n0  = ntb * 16 + col;
        const int n1  = n0 + 16;

        v8f acc[2][4];
        const float bia0 = b2[n0], bia1 = b2[n1];
        #pragma unroll
        for (int ms = 0; ms < 4; ++ms)
            #pragma unroll
            for (int r = 0; r < 8; ++r) { acc[0][ms][r] = bia0; acc[1][ms][r] = bia1; }

        for (int j = 0; j < H_DIM / 32; ++j) {
            // A fragments from LDS: half-wave hi reads H n-tile block 2j+hi (K 16hi..16hi+15)
            v16bf a[4];
            #pragma unroll
            for (int ms = 0; ms < 4; ++ms) {
                const int rb = (16 * ms + col) * 16;
                v8bf lo = *(const v8bf*)(sH + (2 * j + hi) * (MTILE * 16) + rb);
                v8bf hc = *(const v8bf*)(sH + (2 * j + hi) * (MTILE * 16) + rb + 8);
                #pragma unroll
                for (int i = 0; i < 8; ++i) { a[ms][i] = lo[i]; a[ms][i + 8] = hc[i]; }
            }
            v16bf bf0, bf1;
            if (SWZ) {
                const size_t wb = ((size_t)j * 64 + ntb) * 512 + lane * 16;
                v8bf l0 = *(const v8bf*)(swz2 + wb);
                v8bf h0 = *(const v8bf*)(swz2 + wb + 8);
                v8bf l1 = *(const v8bf*)(swz2 + wb + 512);
                v8bf h1 = *(const v8bf*)(swz2 + wb + 520);
                #pragma unroll
                for (int i = 0; i < 8; ++i) {
                    bf0[i] = l0[i]; bf0[i + 8] = h0[i];
                    bf1[i] = l1[i]; bf1[i + 8] = h1[i];
                }
            } else {
                const int kb = 32 * j + 16 * hi;
                #pragma unroll
                for (int i = 0; i < 16; ++i) {
                    const float* wr = W2f + (size_t)(kb + i) * D_DIM;
                    bf0[i] = (bf16_t)wr[n0];
                    bf1[i] = (bf16_t)wr[n1];
                }
            }
            #pragma unroll
            for (int ms = 0; ms < 4; ++ms) {
                acc[0][ms] = wmma_bf16(a[ms], bf0, acc[0][ms]);
                acc[1][ms] = wmma_bf16(a[ms], bf1, acc[1][ms]);
            }
        }

        #pragma unroll
        for (int e = 0; e < 2; ++e)
            #pragma unroll
            for (int ms = 0; ms < 4; ++ms)
                #pragma unroll
                for (int r = 0; r < 8; ++r)
                    Out[(size_t)(m_base + 16 * ms + r + 8 * hi) * D_DIM +
                        (ntb + e) * 16 + col] = acc[e][ms][r];
    }
}

// out[:, 0, :] = hidden_states[:, 0, :]  (8 rows x 1024 f32)
__global__ void copy_first_token_kernel(const float* __restrict__ X,
                                        float* __restrict__ Out) {
    const int i = blockIdx.x * blockDim.x + threadIdx.x;   // 0..8191
    const int b = i >> 10;
    const int d = i & 1023;
    const size_t idx = (size_t)b * S_DIM * D_DIM + d;
    Out[idx] = X[idx];
}

extern "C" void kernel_launch(void* const* d_in, const int* in_sizes, int n_in,
                              void* d_out, int out_size, void* d_ws, size_t ws_size,
                              hipStream_t stream) {
    (void)in_sizes; (void)n_in; (void)out_size;
    const float* X  = (const float*)d_in[0];
    const float* W1 = (const float*)d_in[1];
    const float* b1 = (const float*)d_in[2];
    const float* W2 = (const float*)d_in[3];
    const float* b2 = (const float*)d_in[4];
    float* Out = (float*)d_out;

    const dim3 grid(M_TOT / MTILE);   // 256 workgroups, one 64-token tile each
    const dim3 block(256);            // 8 wave32s

    if (ws_size >= WS_NEEDED && d_ws != nullptr) {
        // Fast path: bf16 weights pre-swizzled into WMMA B-fragment layout.
        bf16_t* swz1 = (bf16_t*)d_ws;
        bf16_t* swz2 = swz1 + W1_ELEMS;
        swizzle_weights_kernel<<<(SWZ1_BLOCKS + SWZ2_BLOCKS) * 32 / 256, 256, 0,
                                 stream>>>(W1, W2, swz1, swz2);

        hipFuncSetAttribute((const void*)mlp_fused_kernel<true>,
                            hipFuncAttributeMaxDynamicSharedMemorySize, LDS_BYTES);
        mlp_fused_kernel<true><<<grid, block, LDS_BYTES, stream>>>(
            X, W1, b1, W2, b2, swz1, swz2, Out);
    } else {
        // Fallback: gather f32 weights directly (convert in-register).
        hipFuncSetAttribute((const void*)mlp_fused_kernel<false>,
                            hipFuncAttributeMaxDynamicSharedMemorySize, LDS_BYTES);
        mlp_fused_kernel<false><<<grid, block, LDS_BYTES, stream>>>(
            X, W1, b1, W2, b2, nullptr, nullptr, Out);
    }

    copy_first_token_kernel<<<(B_DIM * D_DIM) / 256, 256, 0, stream>>>(X, Out);
}
